// SelectiveSSM_1133871366492
// MI455X (gfx1250) — compile-verified, matrix operand order checked by reading, not run on hardware
//
#include <hip/hip_runtime.h>
#include <hip/hip_bf16.h>

// ---------------------------------------------------------------------------
// Selective SSM forward for MI455X (gfx1250, wave32).
//   conv1d -> [GEMM1 bf16 WMMA] -> pointwise discretize -> chunked scan
//          -> [GEMM2 bf16 WMMA] -> out
// GEMMs use V_WMMA_F32_16X16X32_BF16 (fp32 accumulation); scalar math fp32.
// Fragments for A and B operands are contiguous 16-byte loads per lane from
// row-major [row, K] storage (no transposes needed on CDNA5's bf16 layout).
// GEMM1 wave tile 32x80 (2x5 tiles, 1.4 loads/WMMA); GEMM2 wave tile 32x64.
// ---------------------------------------------------------------------------

#define BQ     8
#define LQ     4096
#define DM     1024
#define DS     64
#define NPROJ  320
#define TOK    (BQ * LQ)          // 32768 tokens
#define CHUNK  128
#define NCHUNK (LQ / CHUNK)       // 32
#define EPSF   1.0e-4f

typedef __attribute__((ext_vector_type(16))) __bf16 v16bf;
typedef __attribute__((ext_vector_type(8)))  __bf16 v8bf;
typedef __attribute__((ext_vector_type(8)))  float  v8f;

// float -> bf16 bits, round-to-nearest-even
static __device__ __forceinline__ unsigned short f2bf(float f) {
    unsigned u = __float_as_uint(f);
    u += 0x7FFFu + ((u >> 16) & 1u);
    return (unsigned short)(u >> 16);
}

// Load one 16-element bf16 WMMA fragment: p already includes the
// (lane>=16 ? 8 : 0) K-offset; elements 0..7 = K+0..7, 8..15 = K+16..23.
static __device__ __forceinline__ v16bf load_frag(const unsigned short* p) {
    v8bf lo = *(const v8bf*)p;
    v8bf hi = *(const v8bf*)(p + 16);
    return __builtin_shufflevector(lo, hi, 0,1,2,3,4,5,6,7,8,9,10,11,12,13,14,15);
}

static __device__ __forceinline__ v8f wmma_bf16(v16bf a, v16bf b, v8f c) {
    return __builtin_amdgcn_wmma_f32_16x16x32_bf16(
        /*neg_a=*/false, a, /*neg_b=*/false, b,
        /*c_mod=*/(short)0, c, /*reuse_a=*/false, /*reuse_b=*/false);
}

// C/D layout: VGPR r, lanes 0-15 -> M=r, lanes 16-31 -> M=r+8; N = lane&15.
static __device__ __forceinline__ void store_tile(float* __restrict__ out, int ldo,
                                                  int m, int n, v8f acc,
                                                  const float* __restrict__ bias,
                                                  int lane) {
    const int half = lane >> 4;
    const int col  = lane & 15;
    const float bv = bias[n + col];
    float* po = out + (size_t)(m + half * 8) * ldo + n + col;
#pragma unroll
    for (int r = 0; r < 8; ++r) po[(size_t)r * ldo] = acc[r] + bv;
}

// Same, but non-temporal stores (final output, never re-read: keep L2 clean).
static __device__ __forceinline__ void store_tile_nt(float* __restrict__ out, int ldo,
                                                     int m, int n, v8f acc,
                                                     const float* __restrict__ bias,
                                                     int lane) {
    const int half = lane >> 4;
    const int col  = lane & 15;
    const float bv = bias[n + col];
    float* po = out + (size_t)(m + half * 8) * ldo + n + col;
#pragma unroll
    for (int r = 0; r < 8; ++r) __builtin_nontemporal_store(acc[r] + bv, po + (size_t)r * ldo);
}

// ---------------------------------------------------------------------------
// Kernel 0: convert weights fp32 -> bf16 bits
// ---------------------------------------------------------------------------
__global__ void ssm_cvtw(const float* __restrict__ w1, const float* __restrict__ w2,
                         unsigned short* __restrict__ w1b, unsigned short* __restrict__ w2b) {
    int i = blockIdx.x * blockDim.x + threadIdx.x;
    if (i < NPROJ * DM) w1b[i] = f2bf(w1[i]);
    if (i < DM * DS)    w2b[i] = f2bf(w2[i]);
}

// ---------------------------------------------------------------------------
// Kernel 1: depthwise conv1d (k=4, pad=1, valid length L-1, tail zero-padded),
// emit bf16 activations for GEMM1. xc[t] = sum_j x[t-1+j] * w[j] + b.
// ---------------------------------------------------------------------------
__global__ __launch_bounds__(256)
void ssm_conv(const float* __restrict__ x, const float* __restrict__ cw,
              const float* __restrict__ cb, unsigned short* __restrict__ xcb) {
    size_t idx = (size_t)blockIdx.x * blockDim.x + threadIdx.x;   // TOK*DM threads
    if (idx >= (size_t)TOK * DM) return;
    const int c = (int)(idx & (DM - 1));
    const size_t row = idx >> 10;            // token index b*L + l
    const int l = (int)(row & (LQ - 1));
    float acc = 0.f;
    if (l < LQ - 1) {                        // l == L-1 stays exactly zero (pad)
        const float w0 = cw[c * 4 + 0], w1 = cw[c * 4 + 1];
        const float w2 = cw[c * 4 + 2], w3 = cw[c * 4 + 3];
        const float* px = x + (row << 10) + c;
        if (l >= 1)       acc += px[-(1 << 10)] * w0;   // t-1
        acc += px[0] * w1;                               // t
        acc += px[1 << 10] * w2;                         // t+1  (l <= L-2)
        if (l + 2 <= LQ - 1) acc += px[2 << 10] * w3;    // t+2
        acc += cb[c];
    }
    xcb[idx] = f2bf(acc);
}

// ---------------------------------------------------------------------------
// Kernel 2: GEMM1  proj[32768,320] = xcb[32768,1024] @ W1[320,1024]^T + b1
// Block = 128 threads (4 waves), 64x160 tile; wave = 2x5 WMMA tiles (32x80).
// 14 b128 loads per 10 WMMAs per K-step.
// ---------------------------------------------------------------------------
__global__ __launch_bounds__(128)
void ssm_gemm1(const unsigned short* __restrict__ Abf,   // (TOK,1024)
               const unsigned short* __restrict__ Bbf,   // (320,1024)
               const float* __restrict__ bias,           // (320)
               float* __restrict__ out) {                // (TOK,320)
    const int lane = threadIdx.x & 31;
    const int wave = threadIdx.x >> 5;
    const int m0 = blockIdx.x * 64 + (wave & 1) * 32;
    const int n0 = blockIdx.y * 160 + (wave >> 1) * 80;
    const int half = (lane >> 4) & 1;
    const int lrow = lane & 15;

    v8f acc[2][5];
#pragma unroll
    for (int i = 0; i < 2; ++i)
#pragma unroll
        for (int j = 0; j < 5; ++j) acc[i][j] = (v8f){};

    const unsigned short* pa[2];
    const unsigned short* pb[5];
#pragma unroll
    for (int i = 0; i < 2; ++i)
        pa[i] = Abf + (size_t)(m0 + i * 16 + lrow) * DM + half * 8;
#pragma unroll
    for (int j = 0; j < 5; ++j)
        pb[j] = Bbf + (size_t)(n0 + j * 16 + lrow) * DM + half * 8;

    for (int k0 = 0; k0 < DM; k0 += 32) {
        v16bf a[2], b[5];
#pragma unroll
        for (int i = 0; i < 2; ++i) a[i] = load_frag(pa[i] + k0);
#pragma unroll
        for (int j = 0; j < 5; ++j) b[j] = load_frag(pb[j] + k0);
#pragma unroll
        for (int i = 0; i < 2; ++i)
#pragma unroll
            for (int j = 0; j < 5; ++j)
                acc[i][j] = wmma_bf16(a[i], b[j], acc[i][j]);
    }
#pragma unroll
    for (int i = 0; i < 2; ++i)
#pragma unroll
        for (int j = 0; j < 5; ++j)
            store_tile(out, NPROJ, m0 + i * 16, n0 + j * 16, acc[i][j], bias, lane);
}

// ---------------------------------------------------------------------------
// Kernel 3: pointwise discretization.  Writes A_bar/B_bar as interleaved
// float2 pairs (one b64 load per scan step) plus C_avg.
// ---------------------------------------------------------------------------
__global__ __launch_bounds__(256)
void ssm_pointwise(const float* __restrict__ proj, const float* __restrict__ A_log,
                   float2* __restrict__ ABp, float* __restrict__ Cavg) {
    int idx = blockIdx.x * blockDim.x + threadIdx.x;     // TOK*DS threads
    if (idx >= TOK * DS) return;
    const int s = idx & (DS - 1);
    const int row = idx >> 6;
    const float* pr = proj + (size_t)row * NPROJ;

    const float delta = pr[s];
    const float bavg = 0.5f * (pr[64 + 2 * s] + pr[64 + 2 * s + 1]);
    const float cavg = 0.5f * (pr[192 + 2 * s] + pr[192 + 2 * s + 1]);

    const float A  = -expf(A_log[s]);
    const float sp = fmaxf(delta, 0.f) + log1pf(expf(-fabsf(delta)));   // softplus
    const float dA = sp * A;
    const float Ab = expf(dA);
    float scale;
    if (fabsf(A) < EPSF) {
        scale = 1.f + 0.5f * dA + dA * dA * (1.f / 6.f);                // Taylor
    } else {
        scale = (Ab - 1.f) / A;                                         // A_safe == A here
    }
    ABp[idx] = make_float2(Ab, scale * bavg);
    Cavg[idx] = cavg;
}

// ---------------------------------------------------------------------------
// Chunked scan: h[l] = A_bar[l]*h[l-1] + B_bar[l] per (b,s) chain.
// Pass 1: per-chunk aggregates; Pass 2: carry scan; Pass 3: replay + y=C*h.
// ---------------------------------------------------------------------------
__global__ __launch_bounds__(64)
void ssm_scan1(const float2* __restrict__ ABp,
               float* __restrict__ aggA, float* __restrict__ aggB) {
    const int b = blockIdx.x >> 5;        // NCHUNK == 32
    const int ch = blockIdx.x & 31;
    const int s = threadIdx.x;
    size_t base = ((size_t)b * LQ + (size_t)ch * CHUNK) * DS + s;
    float aA = 1.f, aB = 0.f;
#pragma unroll 4
    for (int t = 0; t < CHUNK; ++t) {
        const float2 p = ABp[base + (size_t)t * DS];
        aB = fmaf(p.x, aB, p.y);
        aA *= p.x;
    }
    const int o = blockIdx.x * DS + s;
    aggA[o] = aA;
    aggB[o] = aB;
}

__global__ __launch_bounds__(64)
void ssm_scan2(const float* __restrict__ aggA, const float* __restrict__ aggB,
               float* __restrict__ carry) {
    const int b = blockIdx.x;
    const int s = threadIdx.x;
    float h = 0.f;
    for (int ch = 0; ch < NCHUNK; ++ch) {
        const int o = (b * NCHUNK + ch) * DS + s;
        carry[o] = h;
        h = fmaf(aggA[o], h, aggB[o]);
    }
}

__global__ __launch_bounds__(64)
void ssm_scan3(const float2* __restrict__ ABp, const float* __restrict__ Cavg,
               const float* __restrict__ carry, unsigned short* __restrict__ yb) {
    const int b = blockIdx.x >> 5;
    const int ch = blockIdx.x & 31;
    const int s = threadIdx.x;
    size_t base = ((size_t)b * LQ + (size_t)ch * CHUNK) * DS + s;
    float h = carry[blockIdx.x * DS + s];
#pragma unroll 4
    for (int t = 0; t < CHUNK; ++t) {
        const size_t o = base + (size_t)t * DS;
        const float2 p = ABp[o];
        h = fmaf(p.x, h, p.y);
        yb[o] = f2bf(Cavg[o] * h);
    }
}

// ---------------------------------------------------------------------------
// Kernel 7: GEMM2  out[32768,1024] = yb[32768,64] @ W2[1024,64]^T + b2
// Block = 128 threads, 64x128 tile; wave = 2x4 WMMA tiles (32x64).
// Non-temporal epilogue stores (output never re-read).
// ---------------------------------------------------------------------------
__global__ __launch_bounds__(128)
void ssm_gemm2(const unsigned short* __restrict__ Abf,   // (TOK,64)
               const unsigned short* __restrict__ Bbf,   // (1024,64)
               const float* __restrict__ bias,           // (1024)
               float* __restrict__ out) {                // (TOK,1024)
    const int lane = threadIdx.x & 31;
    const int wave = threadIdx.x >> 5;
    const int m0 = blockIdx.x * 64 + (wave & 1) * 32;
    const int n0 = blockIdx.y * 128 + (wave >> 1) * 64;
    const int half = (lane >> 4) & 1;
    const int lrow = lane & 15;

    v8f acc[2][4];
#pragma unroll
    for (int i = 0; i < 2; ++i)
#pragma unroll
        for (int j = 0; j < 4; ++j) acc[i][j] = (v8f){};

    const unsigned short* pa[2];
    const unsigned short* pb[4];
#pragma unroll
    for (int i = 0; i < 2; ++i)
        pa[i] = Abf + (size_t)(m0 + i * 16 + lrow) * DS + half * 8;
#pragma unroll
    for (int j = 0; j < 4; ++j)
        pb[j] = Bbf + (size_t)(n0 + j * 16 + lrow) * DS + half * 8;

#pragma unroll
    for (int k0 = 0; k0 < DS; k0 += 32) {
        v16bf a[2], b[4];
#pragma unroll
        for (int i = 0; i < 2; ++i) a[i] = load_frag(pa[i] + k0);
#pragma unroll
        for (int j = 0; j < 4; ++j) b[j] = load_frag(pb[j] + k0);
#pragma unroll
        for (int i = 0; i < 2; ++i)
#pragma unroll
            for (int j = 0; j < 4; ++j)
                acc[i][j] = wmma_bf16(a[i], b[j], acc[i][j]);
    }
#pragma unroll
    for (int i = 0; i < 2; ++i)
#pragma unroll
        for (int j = 0; j < 4; ++j)
            store_tile_nt(out, DM, m0 + i * 16, n0 + j * 16, acc[i][j], bias, lane);
}

// ---------------------------------------------------------------------------
// Workspace layout (bytes).  The xcb region is dead after GEMM1 and is reused
// for AB-pairs/C_avg/y/aggregates.  Total required ~106 MB.
// ---------------------------------------------------------------------------
static const size_t W1B_OFF   = 0;                                   // 655,360
static const size_t W2B_OFF   = 655360;                              // 131,072
static const size_t PROJ_OFF  = 1ull << 20;                          // 41,943,040
static const size_t XCB_OFF   = 42ull << 20;                         // 67,108,864 (bf16 xc)
static const size_t ABP_OFF   = XCB_OFF;                             // 16,777,216 (reuse)
static const size_t CAVG_OFF  = XCB_OFF + 16777216ull;               // 8,388,608
static const size_t YB_OFF    = XCB_OFF + 25165824ull;               // 4,194,304
static const size_t AGGA_OFF  = XCB_OFF + 29360128ull;               // 65,536
static const size_t AGGB_OFF  = AGGA_OFF + 65536ull;
static const size_t CARRY_OFF = AGGB_OFF + 65536ull;

extern "C" void kernel_launch(void* const* d_in, const int* in_sizes, int n_in,
                              void* d_out, int out_size, void* d_ws, size_t ws_size,
                              hipStream_t stream) {
    const float* x    = (const float*)d_in[0];   // (B,L,1024)
    const float* cw   = (const float*)d_in[1];   // (1024,1,4)
    const float* cb   = (const float*)d_in[2];   // (1024)
    const float* w1   = (const float*)d_in[3];   // (320,1024)
    const float* b1   = (const float*)d_in[4];   // (320)
    const float* alog = (const float*)d_in[5];   // (64)
    const float* w2   = (const float*)d_in[6];   // (1024,64)
    const float* b2   = (const float*)d_in[7];   // (1024)
    float* out        = (float*)d_out;           // (B,L,1024)

    char* ws = (char*)d_ws;
    unsigned short* w1b  = (unsigned short*)(ws + W1B_OFF);
    unsigned short* w2b  = (unsigned short*)(ws + W2B_OFF);
    float*          proj = (float*)(ws + PROJ_OFF);
    unsigned short* xcb  = (unsigned short*)(ws + XCB_OFF);
    float2*         ABp  = (float2*)(ws + ABP_OFF);
    float*          Cavg = (float*)(ws + CAVG_OFF);
    unsigned short* yb   = (unsigned short*)(ws + YB_OFF);
    float*          aggA = (float*)(ws + AGGA_OFF);
    float*          aggB = (float*)(ws + AGGB_OFF);
    float*          carr = (float*)(ws + CARRY_OFF);

    // 0) weights -> bf16
    ssm_cvtw<<<(NPROJ * DM + 255) / 256, 256, 0, stream>>>(w1, w2, w1b, w2b);
    // 1) depthwise conv -> bf16 activations
    ssm_conv<<<(TOK * DM) / 256, 256, 0, stream>>>(x, cw, cb, xcb);
    // 2) GEMM1 (bf16 WMMA, fp32 accum), 64x160 block tiles
    ssm_gemm1<<<dim3(TOK / 64, NPROJ / 160), 128, 0, stream>>>(xcb, w1b, b1, proj);
    // 3) discretize -> interleaved (A_bar,B_bar) pairs + C_avg
    ssm_pointwise<<<(TOK * DS) / 256, 256, 0, stream>>>(proj, alog, ABp, Cavg);
    // 4) chunked scan
    ssm_scan1<<<BQ * NCHUNK, DS, 0, stream>>>(ABp, aggA, aggB);
    ssm_scan2<<<BQ, DS, 0, stream>>>(aggA, aggB, carr);
    ssm_scan3<<<BQ * NCHUNK, DS, 0, stream>>>(ABp, Cavg, carr, yb);
    // 5) GEMM2 (bf16 WMMA, fp32 accum), 64x128 block tiles, NT stores
    ssm_gemm2<<<dim3(TOK / 64, DM / 128), 128, 0, stream>>>(yb, w2b, b2, out);
}
